// MultiHeadAtt_36558761624441
// MI455X (gfx1250) — matrix operand order repaired and checked by reading, not compiled, not run on hardware
//
#include <hip/hip_runtime.h>

// Problem constants (match reference)
#define BB 8
#define SS 1024
#define DD 1024
#define HH 16
#define DKK 64

typedef __attribute__((ext_vector_type(16))) __bf16 v16bf;
typedef __attribute__((ext_vector_type(8)))  float  v8f;

// ---------------------------------------------------------------------------
// Fragment loaders per the CDNA5 WMMA VGPR layouts (CDNA5 ISA).
// A: 16x32 MxK. lane&15 = row, lane>>4 = K-half; VGPR0-3 hold K=kh*8+0..7,
// VGPR4-7 hold K=16+kh*8+0..7.
// B: 32x16 KxN sourced from an NxK row-major matrix (all GEMMs are NT).
// lane&15 = n, same K layout as A: VGPR0-3 K=kh*8+0..7, VGPR4-7 K=16+kh*8+0..7.
// ---------------------------------------------------------------------------
__device__ __forceinline__ v16bf load_a_frag(const __bf16* __restrict__ A,
                                             int lda, int lane) {
  const __bf16* p = A + (long)(lane & 15) * lda + ((lane >> 4) << 3);
  union { v16bf v; uint4 q[2]; } u;
  u.q[0] = *(const uint4*)(p);
  u.q[1] = *(const uint4*)(p + 16);
  return u.v;
}

// f32 source variant (used for the softmaxed att matrix): convert on load,
// compiler emits v_cvt_pk_bf16_f32.
__device__ __forceinline__ v16bf load_a_frag(const float* __restrict__ A,
                                             int lda, int lane) {
  const float* p = A + (long)(lane & 15) * lda + ((lane >> 4) << 3);
  v16bf a;
#pragma unroll
  for (int e = 0; e < 8; ++e) a[e] = (__bf16)p[e];
#pragma unroll
  for (int e = 0; e < 8; ++e) a[8 + e] = (__bf16)p[16 + e];
  return a;
}

__device__ __forceinline__ v16bf load_b_frag(const __bf16* __restrict__ Bt,
                                             int ldb, int lane) {
  const __bf16* p = Bt + (long)(lane & 15) * ldb + ((lane >> 4) << 3);
  union { v16bf v; uint4 q[2]; } u;
  u.q[0] = *(const uint4*)(p);
  u.q[1] = *(const uint4*)(p + 16);
  return u.v;
}

// B fragment out of an LDS-staged 64(n) x 64(k) bf16 tile (row-major, 64/row).
__device__ __forceinline__ v16bf load_b_lds(const __bf16* tile, int j, int kk,
                                            int lane) {
  const __bf16* p = tile + (j * 16 + (lane & 15)) * 64 + kk + ((lane >> 4) << 3);
  union { v16bf v; uint4 q[2]; } u;
  u.q[0] = *(const uint4*)(p);
  u.q[1] = *(const uint4*)(p + 16);
  return u.v;
}

__device__ __forceinline__ v8f wmma_bf16(v16bf a, v16bf b, v8f c) {
  return __builtin_amdgcn_wmma_f32_16x16x32_bf16(false, a, false, b,
                                                 (short)0, c, false, false);
}

// ---------------------------------------------------------------------------
// Unified NT GEMM with async-to-LDS double-buffered B staging.
// out[m,n] = sum_k A[m,k] * Bt[n,k] + bias[n]
// 256 threads = 8 waves; wave w owns rows [.. +16) x 64 columns; all waves of
// a block share the same B tile, so it is staged once per 64-wide k-step via
// GLOBAL_LOAD_ASYNC_TO_LDS_B128 (ASYNCcnt) while WMMAs consume the previous
// tile. grid = (N/64, M/128, Z).
// MODE 0: bf16 row-major (ldc)      MODE 1: bf16 Vt[B,H,DKK,SS] layout
// MODE 2: f32 row-major (ldc)       MODE 3: ctx -> cat[:, D:2D] (z = b*H+h)
// ---------------------------------------------------------------------------
template <typename AT, int MODE>
__global__ void gemm_kernel(const AT* __restrict__ A, int lda, long strideA,
                            const __bf16* __restrict__ Bt, int ldb, long strideB,
                            const float* __restrict__ bias,
                            float* __restrict__ outf, __bf16* __restrict__ outb,
                            int ldc, int K) {
  __shared__ __align__(16) __bf16 btile[2][64 * 64];  // 2 x 8KB

  const int lane = threadIdx.x & 31;
  const int wave = threadIdx.x >> 5;
  const int z = blockIdx.z;
  const long m0 = ((long)blockIdx.y * 8 + wave) * 16;
  const int n0 = blockIdx.x * 64;

  const AT* Abase = A + (long)z * strideA + m0 * lda;
  const __bf16* Bbase = Bt + (long)z * strideB + (long)n0 * ldb;

  // Stage one 64x64 bf16 B tile: 512 16-byte chunks, 2 per thread.
  auto stage = [&](int buf, int k) {
#pragma unroll
    for (int i = 0; i < 2; ++i) {
      const int c = (int)threadIdx.x + 256 * i;   // 0..511
      const int row = c >> 3;                     // n row (0..63), 128B each
      const int col = (c & 7) * 8;                // bf16 elements
      const __bf16* g = Bbase + (long)row * ldb + k + col;
      const unsigned lds = (unsigned)(unsigned long long)
                               &btile[buf][row * 64 + col];
      asm volatile("global_load_async_to_lds_b128 %0, %1, off"
                   :: "v"(lds), "v"((unsigned long long)g)
                   : "memory");
    }
  };

  v8f acc[4] = {};
  stage(0, 0);
  int cur = 0;
  for (int k = 0; k < K; k += 64) {
    const bool more = (k + 64) < K;
    if (more) stage(cur ^ 1, k + 64);             // prefetch next tile
    if (more) asm volatile("s_wait_asynccnt 0x2" ::: "memory");
    else      asm volatile("s_wait_asynccnt 0x0" ::: "memory");
    __syncthreads();                              // tile visible to all waves
#pragma unroll
    for (int kk = 0; kk < 64; kk += 32) {
      v16bf a = load_a_frag(Abase + k + kk, lda, lane);
#pragma unroll
      for (int j = 0; j < 4; ++j) {
        v16bf b = load_b_lds(btile[cur], j, kk, lane);
        acc[j] = wmma_bf16(a, b, acc[j]);
      }
    }
    __syncthreads();                              // done reading before reuse
    cur ^= 1;
  }

  const int nl = lane & 15;
  const int kh = lane >> 4;
#pragma unroll
  for (int j = 0; j < 4; ++j) {
    const int n = n0 + j * 16 + nl;
    const float bv = bias ? bias[n] : 0.0f;
#pragma unroll
    for (int r = 0; r < 8; ++r) {
      const long m = m0 + kh * 8 + r;
      const float v = acc[j][r] + bv;
      if (MODE == 0) {
        outb[m * ldc + n] = (__bf16)v;
      } else if (MODE == 1) {
        // Vt[b, h, dkk, s]; m = flat b*S+s row, n = h*64 + dkk
        const long b = m >> 10;
        const long s = m & (SS - 1);
        const int  h = n >> 6;
        const int  dk = n & (DKK - 1);
        outb[(((b * HH + h) * DKK + dk) * (long)SS) + s] = (__bf16)v;
      } else if (MODE == 2) {
        outf[m * ldc + n] = v;
      } else {  // MODE 3: ctx -> cat second half; z = b*H + h, m = seq pos
        const int b = z >> 4;
        const int h = z & 15;
        outb[((long)b * SS + m) * (2 * DD) + DD + h * DKK + n] = (__bf16)v;
      }
    }
  }
}

// ---------------------------------------------------------------------------
// scores[b,h,q,k] = (Qh . Kh) / 8, masked query rows -> -1e9. K = 64,
// fully unrolled, direct bf16 fragment loads. grid = (S/64, S/128, B*H).
// ---------------------------------------------------------------------------
__global__ void scores_kernel(const __bf16* __restrict__ Qp,
                              const __bf16* __restrict__ Kp,
                              const unsigned char* __restrict__ mask,
                              float* __restrict__ att) {
  const int lane = threadIdx.x & 31;
  const int wave = threadIdx.x >> 5;
  const int bh = blockIdx.z;
  const int b = bh >> 4;   // H == 16
  const int h = bh & 15;
  const long m0 = ((long)blockIdx.y * 8 + wave) * 16;
  const int n0 = blockIdx.x * 64;

  const __bf16* Abase = Qp + ((long)b * SS + m0) * DD + h * DKK;
  const __bf16* Bbase = Kp + ((long)b * SS) * DD + h * DKK;

  v8f acc[4] = {};
#pragma unroll
  for (int k = 0; k < DKK; k += 32) {
    v16bf a = load_a_frag(Abase + k, DD, lane);
#pragma unroll
    for (int j = 0; j < 4; ++j) {
      v16bf bf = load_b_frag(Bbase + (long)(n0 + j * 16) * DD + k, DD, lane);
      acc[j] = wmma_bf16(a, bf, acc[j]);
    }
  }

  const int nl = lane & 15;
  const int kh = lane >> 4;
  float* outbh = att + (long)bh * SS * SS;
#pragma unroll
  for (int j = 0; j < 4; ++j) {
    const int n = n0 + j * 16 + nl;
#pragma unroll
    for (int r = 0; r < 8; ++r) {
      const long m = m0 + kh * 8 + r;
      float v = acc[j][r] * 0.125f;                // 1/sqrt(64)
      if (mask[(long)b * SS + m]) v = -1e9f;       // mask whole query row
      outbh[m * SS + n] = v;
    }
  }
}

// ---------------------------------------------------------------------------
// Row softmax over att[B*H*S, S] in place (f32, written to d_out).
// ---------------------------------------------------------------------------
__global__ void softmax_kernel(float* __restrict__ att) {
  __shared__ float red[256];
  float* p = att + (long)blockIdx.x * SS;
  const int t = threadIdx.x;

  float v[4];
  float mx = -3.0e38f;
#pragma unroll
  for (int i = 0; i < 4; ++i) { v[i] = p[t + i * 256]; mx = fmaxf(mx, v[i]); }
  red[t] = mx; __syncthreads();
  for (int s = 128; s > 0; s >>= 1) {
    if (t < s) red[t] = fmaxf(red[t], red[t + s]);
    __syncthreads();
  }
  mx = red[0]; __syncthreads();

  float sum = 0.0f;
#pragma unroll
  for (int i = 0; i < 4; ++i) { v[i] = __expf(v[i] - mx); sum += v[i]; }
  red[t] = sum; __syncthreads();
  for (int s = 128; s > 0; s >>= 1) {
    if (t < s) red[t] += red[t + s];
    __syncthreads();
  }
  const float inv = 1.0f / red[0];
#pragma unroll
  for (int i = 0; i < 4; ++i) p[t + i * 256] = v[i] * inv;
}

// ---------------------------------------------------------------------------
// One-time f32 -> bf16 conversions (8 elements/thread; sizes all divide).
// ---------------------------------------------------------------------------
__global__ void cvt_bf16_kernel(const float* __restrict__ src,
                                __bf16* __restrict__ dst) {
  const long i = ((long)blockIdx.x * blockDim.x + threadIdx.x) * 8;
#pragma unroll
  for (int e = 0; e < 8; ++e) dst[i + e] = (__bf16)src[i + e];
}

// query (f32, [B*S, D]) -> cat[:, 0:D] (bf16, row stride 2D)
__global__ void cvt_query_to_cat(const float* __restrict__ q,
                                 __bf16* __restrict__ cat) {
  const long i = ((long)blockIdx.x * blockDim.x + threadIdx.x) * 8;
  const long m = i >> 10;          // D == 1024
  const int  c = (int)(i & (DD - 1));
#pragma unroll
  for (int e = 0; e < 8; ++e)
    cat[m * (2 * DD) + c + e] = (__bf16)q[i + e];
}

// ---------------------------------------------------------------------------
// Host-side launch
// ---------------------------------------------------------------------------
extern "C" void kernel_launch(void* const* d_in, const int* in_sizes, int n_in,
                              void* d_out, int out_size, void* d_ws,
                              size_t ws_size, hipStream_t stream) {
  (void)in_sizes; (void)n_in; (void)out_size; (void)ws_size;

  const float* key   = (const float*)d_in[0];
  const float* query = (const float*)d_in[1];
  const float* value = (const float*)d_in[2];
  const unsigned char* mask = (const unsigned char*)d_in[3];  // jnp bool_
  const float* Wk = (const float*)d_in[4];
  const float* bk = (const float*)d_in[5];
  const float* Wq = (const float*)d_in[6];
  const float* bq = (const float*)d_in[7];
  const float* Wv = (const float*)d_in[8];
  const float* bv = (const float*)d_in[9];
  const float* Wp = (const float*)d_in[10];
  const float* bp = (const float*)d_in[11];

  // Outputs: att [B*H, S, S] then out [B, S, D], f32, concatenated flat.
  float* att = (float*)d_out;
  float* out = att + (long)BB * HH * SS * SS;

  // Workspace (bf16): ~128 MB total.
  const long N_BSD = (long)BB * SS * DD;        // 8,388,608
  __bf16* cat = (__bf16*)d_ws;                  // [8192, 2048]
  __bf16* kb  = cat + 2 * N_BSD;                // [8192, 1024]
  __bf16* vb  = kb + N_BSD;
  __bf16* Wqb = vb + N_BSD;                     // [1024, 1024]
  __bf16* Wkb = Wqb + (long)DD * DD;
  __bf16* Wvb = Wkb + (long)DD * DD;
  __bf16* Wpb = Wvb + (long)DD * DD;            // [1024, 2048]
  __bf16* Qpb = Wpb + (long)DD * 2 * DD;        // [8192, 1024]
  __bf16* Kpb = Qpb + N_BSD;
  __bf16* Vtb = Kpb + N_BSD;                    // [B, H, 64, S]

  const dim3 blk(256);

  // f32 -> bf16 conversions.
  cvt_bf16_kernel<<<dim3(N_BSD / 2048), blk, 0, stream>>>(key, kb);
  cvt_bf16_kernel<<<dim3(N_BSD / 2048), blk, 0, stream>>>(value, vb);
  cvt_bf16_kernel<<<dim3((DD * DD) / 2048), blk, 0, stream>>>(Wq, Wqb);
  cvt_bf16_kernel<<<dim3((DD * DD) / 2048), blk, 0, stream>>>(Wk, Wkb);
  cvt_bf16_kernel<<<dim3((DD * DD) / 2048), blk, 0, stream>>>(Wv, Wvb);
  cvt_bf16_kernel<<<dim3((DD * 2 * DD) / 2048), blk, 0, stream>>>(Wp, Wpb);
  cvt_query_to_cat<<<dim3(N_BSD / 2048), blk, 0, stream>>>(query, cat);

  // Projections: M = 8192, N = 1024, K = 1024.
  const dim3 gp(DD / 64, (BB * SS) / 128, 1);
  gemm_kernel<__bf16, 0><<<gp, blk, 0, stream>>>(
      cat, 2 * DD, 0, Wqb, DD, 0, bq, nullptr, Qpb, DD, DD);
  gemm_kernel<__bf16, 0><<<gp, blk, 0, stream>>>(
      kb, DD, 0, Wkb, DD, 0, bk, nullptr, Kpb, DD, DD);
  gemm_kernel<__bf16, 1><<<gp, blk, 0, stream>>>(
      vb, DD, 0, Wvb, DD, 0, bv, nullptr, Vtb, 0, DD);

  // Scores + mask, then softmax in place on d_out.
  scores_kernel<<<dim3(SS / 64, SS / 128, BB * HH), blk, 0, stream>>>(
      Qpb, Kpb, mask, att);
  softmax_kernel<<<dim3(BB * HH * SS), blk, 0, stream>>>(att);

  // ctx = att @ V (per head) -> cat[:, D:2D]. A is f32 (softmax output).
  gemm_kernel<float, 3><<<dim3(1, SS / 128, BB * HH), blk, 0, stream>>>(
      att, SS, (long)SS * SS, Vtb, SS, (long)DKK * SS, nullptr, nullptr,
      cat, 0, SS);

  // Output projection: M = 8192, N = 1024, K = 2048, f32 out.
  gemm_kernel<__bf16, 2><<<gp, blk, 0, stream>>>(
      cat, 2 * DD, 0, Wpb, 2 * DD, 0, bp, out, nullptr, DD, 2 * DD);
}
